// AttentionPoolDown_41996190221022
// MI455X (gfx1250) — matrix operand order repaired and checked by reading, not compiled, not run on hardware
//
#include <hip/hip_runtime.h>
#include <hip/hip_bf16.h>
#include <math.h>

// ---------------------------------------------------------------------------
// AttentionPoolDown for MI455X (gfx1250, wave32, WMMA)
//
// Key insight: reference returns out[:, :, :P, :], so only the P=128 pool
// queries matter -> x_q is never read. Memory-bound problem (~39 MB @ 23.3
// TB/s ~ 1.7us), so RoPE is fused into the attention pass (K/V read once),
// and matmuls use v_wmma_f32_16x16x32_f16.
// ---------------------------------------------------------------------------

typedef __attribute__((ext_vector_type(16))) _Float16 v16h;
typedef __attribute__((ext_vector_type(8)))  _Float16 v8h;
typedef __attribute__((ext_vector_type(8)))  float    v8f;

#define B_    2
#define H_    16
#define D_    64
#define T_    2048
#define MAXN  32
#define P_    128          // MAX_N * R
#define L_    (P_ + T_)    // 2176 keys
#define KC    128          // keys per chunk (chunk 0 == pool exactly)
#define NCHUNK (L_ / KC)   // 17

// starts[b][n] = first t with regions[b][t] == n+1, else 0 (argmax semantics)
__global__ void starts_kernel(const int* __restrict__ regions,
                              int* __restrict__ starts) {
  __shared__ int s[MAXN];
  const int b = blockIdx.x;
  const int tid = threadIdx.x;
  if (tid < MAXN) s[tid] = 0x7fffffff;
  __syncthreads();
  for (int t = tid; t < T_; t += blockDim.x) {
    int n = regions[b * T_ + t];
    if (n >= 1 && n <= MAXN) atomicMin(&s[n - 1], t);
  }
  __syncthreads();
  if (tid < MAXN) {
    int v = s[tid];
    starts[b * MAXN + tid] = (v == 0x7fffffff) ? 0 : v;
  }
}

// RoPE on one 32-dim half: pairs (j, j+16), inv_freq[j] = theta^(-j/16)
__device__ __forceinline__ void rope32(const float* __restrict__ src,
                                       float pos, _Float16* __restrict__ dst) {
  const float kf = -0.5756462732485114f; // -ln(10000)/16
#pragma unroll
  for (int j = 0; j < 16; ++j) {
    float ang = pos * __expf(kf * (float)j);
    float sn, cs;
    __sincosf(ang, &sn, &cs);
    float x1 = src[j], x2 = src[j + 16];
    dst[j]      = (_Float16)(x1 * cs - x2 * sn);
    dst[j + 16] = (_Float16)(x1 * sn + x2 * cs);
  }
}

__global__ __launch_bounds__(256)
void attn_kernel(const float* __restrict__ pool_q,
                 const float* __restrict__ pool_k,
                 const float* __restrict__ pool_v,
                 const float* __restrict__ x_k,
                 const float* __restrict__ x_v,
                 const float* __restrict__ slopes,
                 const int*   __restrict__ regions,
                 const unsigned char* __restrict__ t_mask,
                 const unsigned char* __restrict__ n_mask,
                 const int*   __restrict__ starts,
                 float* __restrict__ out) {
  __shared__ __align__(16) _Float16 Qs[P_][D_];       // 16 KB  roped Q (f16)
  __shared__ __align__(16) _Float16 Ks[KC][D_];       // 16 KB  roped K chunk
  __shared__ __align__(16) _Float16 Vt[D_][KC];       // 16 KB  V chunk, transposed
  __shared__ __align__(16) _Float16 Ps[8][16][KC];    // 32 KB  per-wave P staging
  __shared__ float Rk[KC];                            // region idx (+inf = masked)

  const int bh   = blockIdx.x;
  const int b    = bh / H_;
  const int h    = bh % H_;
  const int tid  = threadIdx.x;
  const int wave = tid >> 5;
  const int lane = tid & 31;
  const int lhi  = (lane >= 16) ? 1 : 0;
  const int l16  = lane & 15;
  const float slope = slopes[h];
  const float scale = 0.125f; // 1/sqrt(64)

  // ---- Phase 0: RoPE all 128 pool queries into LDS (thread = (row, half)) --
  {
    const int qr = tid >> 1, half = tid & 1;
    const int n  = qr >> 2;
    const float pos = (half == 0)
        ? (float)starts[b * MAXN + n] + 0.5f * (float)(qr & 3)   // gpos
        : (float)(n + 1);                                        // region idx
    const float* src = pool_q + (((size_t)(b * H_ + h)) * P_ + qr) * D_ + 32 * half;
    float buf[32];
#pragma unroll
    for (int i = 0; i < 8; ++i) ((float4*)buf)[i] = ((const float4*)src)[i];
    _Float16 d[32];
    rope32(buf, pos, d);
#pragma unroll
    for (int i = 0; i < 32; ++i) Qs[qr][32 * half + i] = d[i];
  }
  __syncthreads();

  // ---- Per-wave Q A-frags (16x32 f16 layout per ISA 7.12.2) ----------------
  const int qbase = wave * 16;
  const int qrow  = qbase + l16;
  v16h aQ[2];
#pragma unroll
  for (int ks = 0; ks < 2; ++ks) {
    const int dbase = 32 * ks + 8 * lhi; // lanes<16: K {0..7,16..23}; else {8..15,24..31}
    v8h lo = *(const v8h*)&Qs[qrow][dbase];
    v8h hi = *(const v8h*)&Qs[qrow][dbase + 16];
#pragma unroll
    for (int i = 0; i < 8; ++i) { aQ[ks][i] = lo[i]; aQ[ks][i + 8] = hi[i]; }
  }

  // ---- Flash-attention state (C-layout: VGPR j <-> row j + 8*lhi) ----------
  float rowm[8], rowl[8], rq[8];
  v8f   acc[4] = {};
#pragma unroll
  for (int j = 0; j < 8; ++j) {
    rowm[j] = -INFINITY;
    rowl[j] = 0.f;
    rq[j]   = (float)((qbase + j + 8 * lhi) / 4 + 1); // query region idx
  }

  for (int chunk = 0; chunk < NCHUNK; ++chunk) {
    __syncthreads();
    // ---- Load + RoPE K, transpose V into LDS (thread = (key, half)) --------
    {
      const int kk = tid >> 1, half = tid & 1;
      const int kidx = chunk * KC + kk;
      float gpos, rix; int valid;
      const float *ksrc, *vsrc;
      if (kidx < P_) {
        const int n = kidx >> 2;
        valid = n_mask[b * MAXN + n] != 0;
        gpos  = (float)starts[b * MAXN + n] + 0.5f * (float)(kidx & 3);
        rix   = (float)(n + 1);
        ksrc  = pool_k + (((size_t)(b * H_ + h)) * P_ + kidx) * D_;
        vsrc  = pool_v + (((size_t)(b * H_ + h)) * P_ + kidx) * D_;
      } else {
        const int t = kidx - P_;
        valid = t_mask[b * T_ + t] != 0;
        gpos  = (float)t;
        rix   = (float)regions[b * T_ + t];
        ksrc  = x_k + (((size_t)(b * H_ + h)) * T_ + t) * D_;
        vsrc  = x_v + (((size_t)(b * H_ + h)) * T_ + t) * D_;
      }
      float buf[32];
#pragma unroll
      for (int i = 0; i < 8; ++i) ((float4*)buf)[i] = ((const float4*)(ksrc + 32 * half))[i];
      _Float16 dk[32];
      rope32(buf, half == 0 ? gpos : rix, dk);
#pragma unroll
      for (int i = 0; i < 32; ++i) Ks[kk][32 * half + i] = dk[i];
#pragma unroll
      for (int i = 0; i < 8; ++i) ((float4*)buf)[i] = ((const float4*)(vsrc + 32 * half))[i];
#pragma unroll
      for (int i = 0; i < 32; ++i) Vt[32 * half + i][kk] = (_Float16)buf[i];
      if (half == 0) Rk[kk] = valid ? rix : INFINITY; // inf => bias = -inf
    }
    __syncthreads();

    // Prefetch next chunk's K/V cachelines (global_prefetch_b8)
    if (chunk + 1 < NCHUNK) {
      const int kk = tid >> 1, half = tid & 1;
      const int t = (chunk + 1) * KC + kk - P_;
      if (t >= 0) {
        __builtin_prefetch(x_k + (((size_t)(b * H_ + h)) * T_ + t) * D_ + 32 * half, 0, 0);
        __builtin_prefetch(x_v + (((size_t)(b * H_ + h)) * T_ + t) * D_ + 32 * half, 0, 0);
      }
    }

    // ---- S = Q . K^T  (8 key-tiles x 2 k-steps of WMMA f16) ----------------
    v8f S[8];
#pragma unroll
    for (int nt = 0; nt < 8; ++nt) {
      v8f c = {};
#pragma unroll
      for (int ks = 0; ks < 2; ++ks) {
        const int key   = nt * 16 + l16;          // B: N = lane%16
        const int dbase = 32 * ks + 8 * lhi;      // B: lanes<16: K {0..7,16..23}; else {8..15,24..31}
        v8h lo = *(const v8h*)&Ks[key][dbase];
        v8h hi = *(const v8h*)&Ks[key][dbase + 16];
        v16h bK;
#pragma unroll
        for (int i = 0; i < 8; ++i) { bK[i] = lo[i]; bK[i + 8] = hi[i]; }
        c = __builtin_amdgcn_wmma_f32_16x16x32_f16(false, aQ[ks], false, bK,
                                                   (short)0, c, false, false);
      }
      S[nt] = c;
    }

    // ---- logits = S*scale - slope*|rq - rk|  (masked key: rk=inf -> -inf) --
#pragma unroll
    for (int nt = 0; nt < 8; ++nt) {
      const float rk = Rk[nt * 16 + l16];
#pragma unroll
      for (int j = 0; j < 8; ++j)
        S[nt][j] = S[nt][j] * scale - slope * fabsf(rq[j] - rk);
    }

    // ---- online softmax: row max via 16-lane butterfly ---------------------
    float mnew[8], alpha[8];
#pragma unroll
    for (int j = 0; j < 8; ++j) {
      float m = S[0][j];
#pragma unroll
      for (int nt = 1; nt < 8; ++nt) m = fmaxf(m, S[nt][j]);
#pragma unroll
      for (int off = 1; off < 16; off <<= 1) m = fmaxf(m, __shfl_xor(m, off, 32));
      mnew[j]  = fmaxf(rowm[j], m);
      alpha[j] = (mnew[j] == -INFINITY) ? 0.f : __expf(rowm[j] - mnew[j]);
      rowm[j]  = mnew[j];
    }

    // ---- P = exp(S - mnew); row sums; stage P in LDS (C- -> A-layout) ------
    float rs[8];
#pragma unroll
    for (int j = 0; j < 8; ++j) rs[j] = 0.f;
#pragma unroll
    for (int nt = 0; nt < 8; ++nt) {
#pragma unroll
      for (int j = 0; j < 8; ++j) {
        float p = (mnew[j] == -INFINITY) ? 0.f : __expf(S[nt][j] - mnew[j]);
        rs[j] += p;
        Ps[wave][j + 8 * lhi][nt * 16 + l16] = (_Float16)p;
      }
    }
#pragma unroll
    for (int j = 0; j < 8; ++j) {
      float s = rs[j];
#pragma unroll
      for (int off = 1; off < 16; off <<= 1) s += __shfl_xor(s, off, 32);
      rowl[j] = rowl[j] * alpha[j] + s;
    }
#pragma unroll
    for (int v = 0; v < 4; ++v)
#pragma unroll
      for (int j = 0; j < 8; ++j) acc[v][j] *= alpha[j];

    // ---- O += P . V  (4 k-steps of 32 keys x 4 dim-tiles) ------------------
#pragma unroll
    for (int ks = 0; ks < 4; ++ks) {
      v16h aP;
      {
        const int kb = 32 * ks + 8 * lhi;
        v8h lo = *(const v8h*)&Ps[wave][l16][kb];
        v8h hi = *(const v8h*)&Ps[wave][l16][kb + 16];
#pragma unroll
        for (int i = 0; i < 8; ++i) { aP[i] = lo[i]; aP[i + 8] = hi[i]; }
      }
#pragma unroll
      for (int v = 0; v < 4; ++v) {
        const int dim = v * 16 + l16;
        const int kb  = 32 * ks + 8 * lhi;
        v8h lo = *(const v8h*)&Vt[dim][kb];
        v8h hi = *(const v8h*)&Vt[dim][kb + 16];
        v16h bV;
#pragma unroll
        for (int i = 0; i < 8; ++i) { bV[i] = lo[i]; bV[i + 8] = hi[i]; }
        acc[v] = __builtin_amdgcn_wmma_f32_16x16x32_f16(false, aP, false, bV,
                                                        (short)0, acc[v], false, false);
      }
    }
  }

  // ---- normalize and store -------------------------------------------------
#pragma unroll
  for (int j = 0; j < 8; ++j) rowl[j] = (rowl[j] > 0.f) ? 1.f / rowl[j] : 0.f;
#pragma unroll
  for (int v = 0; v < 4; ++v) {
#pragma unroll
    for (int j = 0; j < 8; ++j) {
      const int row = qbase + j + 8 * lhi;
      const int dim = v * 16 + l16;
      out[(((size_t)(b * H_ + h)) * P_ + row) * D_ + dim] = acc[v][j] * rowl[j];
    }
  }
}

extern "C" void kernel_launch(void* const* d_in, const int* in_sizes, int n_in,
                              void* d_out, int out_size, void* d_ws, size_t ws_size,
                              hipStream_t stream) {
  const float* pool_q = (const float*)d_in[0];
  const float* pool_k = (const float*)d_in[1];
  const float* pool_v = (const float*)d_in[2];
  // d_in[3] == x_q: intentionally unused (output slices to pool queries only)
  const float* x_k    = (const float*)d_in[4];
  const float* x_v    = (const float*)d_in[5];
  const float* slopes = (const float*)d_in[6];
  const int*   regions = (const int*)d_in[7];
  const unsigned char* t_mask = (const unsigned char*)d_in[8];
  const unsigned char* n_mask = (const unsigned char*)d_in[9];
  int* starts = (int*)d_ws; // B*MAXN ints

  starts_kernel<<<B_, 256, 0, stream>>>(regions, starts);
  attn_kernel<<<B_ * H_, 256, 0, stream>>>(pool_q, pool_k, pool_v, x_k, x_v,
                                           slopes, regions, t_mask, n_mask,
                                           starts, (float*)d_out);
}
